// MultiHeadAttention_7524782703116
// MI455X (gfx1250) — matrix operand order repaired and checked by reading, not compiled, not run on hardware
//
#include <hip/hip_runtime.h>
#include <hip/hip_bf16.h>

// ---------------- problem dims ----------------
#define SEQ   2048
#define DIM   1024
#define NH    16
#define HD    64
#define NB    2
#define NTOK  (NB*SEQ)          // 4096
#define SPAD  (SEQ + 4)         // padded LDS row stride (floats) for attention
#define KPAD  40                // padded LDS row stride (halves) for GEMM slabs (80B, 16B-aligned)

typedef __attribute__((ext_vector_type(16))) _Float16 v16h;
typedef __attribute__((ext_vector_type(8)))  float    v8f;

union Frag { v16h v; uint4 q[2]; };

static __device__ __forceinline__ v8f wmma_f16(v16h a, v16h b, v8f c) {
  // D = A(16x32 f16) * B(32x16 f16) + C(16x16 f32)
  return __builtin_amdgcn_wmma_f32_16x16x32_f16(false, a, false, b, (short)0, c,
                                                false, false);
}

// Async global->LDS 16-byte copy (CDNA5, tracked by ASYNCcnt).
static __device__ __forceinline__ void async_ld16(const _Float16* g, unsigned ldsByteOff) {
  unsigned long long ga = (unsigned long long)(uintptr_t)g;
  asm volatile("global_load_async_to_lds_b128 %0, %1, off"
               :: "v"(ldsByteOff), "v"(ga) : "memory");
}
static __device__ __forceinline__ void wait_async_le3() {
  asm volatile("s_wait_asynccnt 0x3" ::: "memory");
}
static __device__ __forceinline__ void wait_async_0() {
  asm volatile("s_wait_asynccnt 0x0" ::: "memory");
}

// ---------------- f32 -> f16 convert (4-wide) ----------------
__global__ void cvt_f32_f16(const float* __restrict__ s, _Float16* __restrict__ d, int n4) {
  int i = blockIdx.x * 256 + threadIdx.x;
  if (i < n4) {
    float4 f = ((const float4*)s)[i];
    union { _Float16 h[4]; uint2 u; } p;
    p.h[0] = (_Float16)f.x; p.h[1] = (_Float16)f.y;
    p.h[2] = (_Float16)f.z; p.h[3] = (_Float16)f.w;
    ((uint2*)d)[i] = p.u;
  }
}

// ---------------- WMMA GEMM: C[M,N] = A[M,K] @ W[N,K]^T ----------------
// M = NTOK = 4096, N = K = DIM = 1024.
// Workgroup: 8 waves (4x2), each wave a 32x32 tile -> 128x64 per block.
// K=32 slabs of A(128x32) and W(64x32) are double-buffered in LDS via
// async loads; fragments are then built with ds_load_b128.
// MODE 0: f32 out, plain [M][N];  MODE 1: f16 [B,H,S,HD];  MODE 2: f16 [B,H,HD,S].
template <int MODE>
__device__ __forceinline__ void store_tile(const v8f& c, int mTile, int nTile,
                                           int lhi, int l15,
                                           float* outF32, _Float16* outF16) {
#pragma unroll
  for (int r = 0; r < 8; r++) {
    int m = mTile + lhi * 8 + r;       // C layout: lanes16-31 hold M+8
    int n = nTile + l15;
    float val = c[r];
    if (MODE == 0) {
      outF32[(size_t)m * DIM + n] = val;
    } else {
      int b = m >> 11, s = m & (SEQ - 1);
      int h = n >> 6,  dh = n & (HD - 1);
      if (MODE == 1)
        outF16[(((size_t)(b * NH + h)) * SEQ + s) * HD + dh] = (_Float16)val;
      else
        outF16[(((size_t)(b * NH + h)) * HD + dh) * SEQ + s] = (_Float16)val;
    }
  }
}

template <int MODE>
__global__ __launch_bounds__(256) void gemm_f16(const _Float16* __restrict__ A,
                                                const _Float16* __restrict__ W,
                                                float* __restrict__ outF32,
                                                _Float16* __restrict__ outF16) {
  __shared__ __align__(16) _Float16 Abuf[2][128 * KPAD];   // 2 x 10240 B
  __shared__ __align__(16) _Float16 Bbuf[2][64 * KPAD];    // 2 x  5120 B

  const int tid  = threadIdx.x;
  const int lane = tid & 31;
  const int w    = tid >> 5;             // 0..7
  const int wm   = w >> 1, wn = w & 1;   // 4 x 2 wave grid
  const int l15  = lane & 15, lhi = lane >> 4;
  const int mBlock = blockIdx.x * 128;
  const int nBlock = blockIdx.y * 64;

  const unsigned aBase0 = (unsigned)(uintptr_t)(void*)&Abuf[0][0];
  const unsigned aBase1 = (unsigned)(uintptr_t)(void*)&Abuf[1][0];
  const unsigned bBase0 = (unsigned)(uintptr_t)(void*)&Bbuf[0][0];
  const unsigned bBase1 = (unsigned)(uintptr_t)(void*)&Bbuf[1][0];

  // Per-thread async-copy assignment:
  //   A slab: 128 rows x 4 chunks(16B) = 512 chunks -> 2 per thread
  //   B slab:  64 rows x 4 chunks(16B) = 256 chunks -> 1 per thread
  const int arow0 = tid >> 2,          ach0 = tid & 3;
  const int arow1 = (tid + 256) >> 2,  ach1 = tid & 3;
  const int brow  = tid >> 2,          bch  = tid & 3;
  const _Float16* aG0 = A + (size_t)(mBlock + arow0) * DIM + ach0 * 8;
  const _Float16* aG1 = A + (size_t)(mBlock + arow1) * DIM + ach1 * 8;
  const _Float16* bG  = W + (size_t)(nBlock + brow)  * DIM + bch  * 8;
  const unsigned aL0 = (unsigned)(arow0 * KPAD + ach0 * 8) * 2u;
  const unsigned aL1 = (unsigned)(arow1 * KPAD + ach1 * 8) * 2u;
  const unsigned bL  = (unsigned)(brow  * KPAD + bch  * 8) * 2u;

  v8f c00 = {}, c01 = {}, c10 = {}, c11 = {};

  // prologue: slab 0 -> buffer 0
  async_ld16(aG0, aBase0 + aL0);
  async_ld16(aG1, aBase0 + aL1);
  async_ld16(bG,  bBase0 + bL);

  const int NSTEP = DIM / 32;                  // 32 k-steps
  for (int s = 0; s < NSTEP; s++) {
    if (s + 1 < NSTEP) {                       // stream next slab while computing
      const int k1 = (s + 1) * 32;
      const unsigned aB = ((s + 1) & 1) ? aBase1 : aBase0;
      const unsigned bB = ((s + 1) & 1) ? bBase1 : bBase0;
      async_ld16(aG0 + k1, aB + aL0);
      async_ld16(aG1 + k1, aB + aL1);
      async_ld16(bG  + k1, bB + bL);
      wait_async_le3();                        // current slab's 3 ops done
    } else {
      wait_async_0();
    }
    __syncthreads();                           // slab s visible to all waves

    const _Float16* As = &Abuf[s & 1][0];
    const _Float16* Bs = &Bbuf[s & 1][0];
    Frag a0, a1, b0, b1;
    const int ar0 = (wm * 32 +      l15) * KPAD;
    const int ar1 = (wm * 32 + 16 + l15) * KPAD;
    const int br0 = (wn * 32 +      l15) * KPAD;
    const int br1 = (wn * 32 + 16 + l15) * KPAD;
    a0.q[0] = *(const uint4*)(As + ar0 +      lhi * 8);
    a0.q[1] = *(const uint4*)(As + ar0 + 16 + lhi * 8);
    a1.q[0] = *(const uint4*)(As + ar1 +      lhi * 8);
    a1.q[1] = *(const uint4*)(As + ar1 + 16 + lhi * 8);
    b0.q[0] = *(const uint4*)(Bs + br0 +      lhi * 8);
    b0.q[1] = *(const uint4*)(Bs + br0 + 16 + lhi * 8);
    b1.q[0] = *(const uint4*)(Bs + br1 +      lhi * 8);
    b1.q[1] = *(const uint4*)(Bs + br1 + 16 + lhi * 8);

    c00 = wmma_f16(a0.v, b0.v, c00);
    c01 = wmma_f16(a0.v, b1.v, c01);
    c10 = wmma_f16(a1.v, b0.v, c10);
    c11 = wmma_f16(a1.v, b1.v, c11);
    __syncthreads();                           // done reading before overwrite
  }

  store_tile<MODE>(c00, mBlock + wm * 32,      nBlock + wn * 32,      lhi, l15, outF32, outF16);
  store_tile<MODE>(c01, mBlock + wm * 32,      nBlock + wn * 32 + 16, lhi, l15, outF32, outF16);
  store_tile<MODE>(c10, mBlock + wm * 32 + 16, nBlock + wn * 32,      lhi, l15, outF32, outF16);
  store_tile<MODE>(c11, mBlock + wm * 32 + 16, nBlock + wn * 32 + 16, lhi, l15, outF32, outF16);
}

// ---------------- fused attention ----------------
// One workgroup (8 waves) handles a 16-query block of one (b,h):
//   scores(16x2048) via WMMA -> LDS, mask+softmax in LDS, attn -> HBM once,
//   attn@V via WMMA (split-K across wave pairs) -> f16 head output.
__global__ __launch_bounds__(256) void attn_fused(const _Float16* __restrict__ qh,
                                                  const _Float16* __restrict__ kh,
                                                  const _Float16* __restrict__ vhT,
                                                  const int* __restrict__ mask,
                                                  float* __restrict__ attnOut,
                                                  _Float16* __restrict__ headOut) {
  extern __shared__ float smem[];
  float* sc   = smem;                    // [16][SPAD]
  float* red  = smem + 16 * SPAD;        // [16][16]
  float* part = red + 256;               // [64][16] split-K partials (4 n-tiles)

  const int tid  = threadIdx.x;
  const int lane = tid & 31, w = tid >> 5;
  const int l15 = lane & 15, lhi = lane >> 4;

  const int q0 = blockIdx.x * 16;
  const int h  = blockIdx.y, b = blockIdx.z;
  const int bh = b * NH + h;

  // ---- phase 1: scores = (qh @ kh^T), 128 k-tiles split over 8 waves ----
  const _Float16* qrow = qh + ((size_t)bh * SEQ + q0 + l15) * HD;
  Frag aq0, aq1;                                    // HD=64 -> two K=32 steps
  aq0.q[0] = *(const uint4*)(qrow +      lhi * 8);
  aq0.q[1] = *(const uint4*)(qrow + 16 + lhi * 8);
  aq1.q[0] = *(const uint4*)(qrow + 32 + lhi * 8);
  aq1.q[1] = *(const uint4*)(qrow + 48 + lhi * 8);

  for (int t = 0; t < 16; t++) {
    const int kt = w * 16 + t;                      // key tile 0..127
    const _Float16* krow = kh + ((size_t)bh * SEQ + kt * 16 + l15) * HD;
    Frag bk;
    v8f c = {};
    bk.q[0] = *(const uint4*)(krow + lhi * 8);
    bk.q[1] = *(const uint4*)(krow + 16 + lhi * 8);
    c = wmma_f16(aq0.v, bk.v, c);
    bk.q[0] = *(const uint4*)(krow + 32 + lhi * 8);
    bk.q[1] = *(const uint4*)(krow + 48 + lhi * 8);
    c = wmma_f16(aq1.v, bk.v, c);
#pragma unroll
    for (int r = 0; r < 8; r++)
      sc[(size_t)(lhi * 8 + r) * SPAD + kt * 16 + l15] = c[r];
  }
  __syncthreads();

  // ---- phase 2: mask + softmax (16 threads per query row) ----
  const int row = tid >> 4;            // 0..15
  const int sub = tid & 15;
  const int qg  = q0 + row;
  float mmax = -3.0e38f;
  for (int j = 0; j < SEQ / 16; j++) {
    int cc = sub + j * 16;
    float vv = sc[row * SPAD + cc] * 0.125f;              // 1/sqrt(HD)
    if (mask[(size_t)qg * SEQ + cc] == 0) vv = -1.0e34f;
    sc[row * SPAD + cc] = vv;
    mmax = fmaxf(mmax, vv);
  }
  red[row * 16 + sub] = mmax;
  __syncthreads();
  if (sub == 0) {
    float m2 = red[row * 16];
    for (int j = 1; j < 16; j++) m2 = fmaxf(m2, red[row * 16 + j]);
    red[row * 16] = m2;
  }
  __syncthreads();
  const float rowMax = red[row * 16];
  float ssum = 0.f;
  for (int j = 0; j < SEQ / 16; j++) {
    int cc = sub + j * 16;
    float e = __expf(sc[row * SPAD + cc] - rowMax);
    sc[row * SPAD + cc] = e;
    ssum += e;
  }
  __syncthreads();
  red[row * 16 + sub] = ssum;
  __syncthreads();
  if (sub == 0) {
    float s2 = 0.f;
    for (int j = 0; j < 16; j++) s2 += red[row * 16 + j];
    red[row * 16] = s2;
  }
  __syncthreads();
  const float inv = 1.0f / red[row * 16];
  for (int j = 0; j < SEQ / 16; j++) sc[row * SPAD + sub + j * 16] *= inv;
  __syncthreads();

  // ---- attn written to HBM exactly once (fully coalesced block copy) ----
  float* dstAttn = attnOut + ((size_t)bh * SEQ + q0) * SEQ;
  for (int e = tid; e < 16 * SEQ; e += 256) {
    int r = e >> 11, cc = e & (SEQ - 1);
    dstAttn[e] = sc[r * SPAD + cc];
  }

  // ---- phase 3: out = attn @ V  (4 n-tiles x split-K over wave pairs) ----
  const int nt = w & 3, khalf = w >> 2;
  const _Float16* vrow = vhT + ((size_t)bh * HD + nt * 16 + l15) * SEQ;
  v8f c = {};
  for (int ks = khalf * 32; ks < khalf * 32 + 32; ks++) {   // 64 K=32 steps total
    const int k0 = ks * 32;
    const float* arow = sc + (size_t)l15 * SPAD + k0 + lhi * 8;
    float4 t0 = *(const float4*)(arow);
    float4 t1 = *(const float4*)(arow + 4);
    float4 t2 = *(const float4*)(arow + 16);
    float4 t3 = *(const float4*)(arow + 20);
    v16h av;
    av[0]  = (_Float16)t0.x; av[1]  = (_Float16)t0.y; av[2]  = (_Float16)t0.z; av[3]  = (_Float16)t0.w;
    av[4]  = (_Float16)t1.x; av[5]  = (_Float16)t1.y; av[6]  = (_Float16)t1.z; av[7]  = (_Float16)t1.w;
    av[8]  = (_Float16)t2.x; av[9]  = (_Float16)t2.y; av[10] = (_Float16)t2.z; av[11] = (_Float16)t2.w;
    av[12] = (_Float16)t3.x; av[13] = (_Float16)t3.y; av[14] = (_Float16)t3.z; av[15] = (_Float16)t3.w;
    Frag bv;
    bv.q[0] = *(const uint4*)(vrow + k0 + lhi * 8);
    bv.q[1] = *(const uint4*)(vrow + k0 + 16 + lhi * 8);
    c = wmma_f16(av, bv.v, c);
  }
  if (khalf == 1) {
#pragma unroll
    for (int r = 0; r < 8; r++)
      part[(nt * 16 + lhi * 8 + r) * 16 + l15] = c[r];
  }
  __syncthreads();
  if (khalf == 0) {
#pragma unroll
    for (int r = 0; r < 8; r++) {
      float val = c[r] + part[(nt * 16 + lhi * 8 + r) * 16 + l15];
      int m = lhi * 8 + r;                            // query within block
      headOut[(size_t)(b * SEQ + q0 + m) * DIM + h * HD + nt * 16 + l15] =
          (_Float16)val;
    }
  }
}

// ---------------- host-side launch ----------------
extern "C" void kernel_launch(void* const* d_in, const int* in_sizes, int n_in,
                              void* d_out, int out_size, void* d_ws, size_t ws_size,
                              hipStream_t stream) {
  (void)in_sizes; (void)n_in; (void)out_size; (void)ws_size;
  const float* q    = (const float*)d_in[0];
  const float* k    = (const float*)d_in[1];
  const float* v    = (const float*)d_in[2];
  const int*   mask = (const int*)  d_in[3];
  const float* Wq   = (const float*)d_in[4];
  const float* Wk   = (const float*)d_in[5];
  const float* Wv   = (const float*)d_in[6];
  const float* Wo   = (const float*)d_in[7];

  char* ws = (char*)d_ws;
  const size_t SZ_X = (size_t)NTOK * DIM * 2;   // 8 MiB f16
  const size_t SZ_W = (size_t)DIM * DIM * 2;    // 2 MiB f16
  _Float16* xq16  = (_Float16*)(ws);
  _Float16* xk16  = (_Float16*)(ws + SZ_X);
  _Float16* xv16  = (_Float16*)(ws + 2 * SZ_X);
  _Float16* wq16  = (_Float16*)(ws + 3 * SZ_X);
  _Float16* wk16  = (_Float16*)(ws + 3 * SZ_X + SZ_W);
  _Float16* wv16  = (_Float16*)(ws + 3 * SZ_X + 2 * SZ_W);
  _Float16* wo16  = (_Float16*)(ws + 3 * SZ_X + 3 * SZ_W);
  _Float16* qhB   = (_Float16*)(ws + 3 * SZ_X + 4 * SZ_W);
  _Float16* khB   = (_Float16*)(ws + 4 * SZ_X + 4 * SZ_W);
  _Float16* vhT   = (_Float16*)(ws + 5 * SZ_X + 4 * SZ_W);
  _Float16* hOut  = (_Float16*)(ws + 6 * SZ_X + 4 * SZ_W);   // ends at 64 MiB

  const int nX4 = NTOK * DIM / 4, nW4 = DIM * DIM / 4;
  cvt_f32_f16<<<(nX4 + 255) / 256, 256, 0, stream>>>(q,  xq16, nX4);
  cvt_f32_f16<<<(nX4 + 255) / 256, 256, 0, stream>>>(k,  xk16, nX4);
  cvt_f32_f16<<<(nX4 + 255) / 256, 256, 0, stream>>>(v,  xv16, nX4);
  cvt_f32_f16<<<(nW4 + 255) / 256, 256, 0, stream>>>(Wq, wq16, nW4);
  cvt_f32_f16<<<(nW4 + 255) / 256, 256, 0, stream>>>(Wk, wk16, nW4);
  cvt_f32_f16<<<(nW4 + 255) / 256, 256, 0, stream>>>(Wv, wv16, nW4);
  cvt_f32_f16<<<(nW4 + 255) / 256, 256, 0, stream>>>(Wo, wo16, nW4);

  dim3 ggrid(NTOK / 128, DIM / 64);  // 32 x 16
  gemm_f16<1><<<ggrid, 256, 0, stream>>>(xq16, wq16, nullptr, qhB);
  gemm_f16<1><<<ggrid, 256, 0, stream>>>(xk16, wk16, nullptr, khB);
  gemm_f16<2><<<ggrid, 256, 0, stream>>>(xv16, wv16, nullptr, vhT);

  float* outPtr  = (float*)d_out;
  float* attnPtr = outPtr + (size_t)NTOK * DIM;     // tuple: (out, attn)
  const size_t smemBytes = (size_t)(16 * SPAD + 256 + 1024) * sizeof(float);
  attn_fused<<<dim3(SEQ / 16, NH, NB), 256, smemBytes, stream>>>(
      qhB, khB, vhT, mask, attnPtr, hOut);

  gemm_f16<0><<<ggrid, 256, 0, stream>>>(hOut, wo16, outPtr, nullptr);
}
